// RWKV_7_39127152066665
// MI455X (gfx1250) — matrix operand order repaired and checked by reading, not compiled, not run on hardware
//
#include <hip/hip_runtime.h>
#include <hip/hip_bf16.h>
#include <stdint.h>
#include <stddef.h>

// ---------------------------------------------------------------------------
// RWKV-7 MoE forward for MI455X (gfx1250, wave32).
// Heavy GEMMs: v_wmma_f32_16x16x32_bf16 (fp32 accumulate), double-buffered LDS
// staged with GLOBAL_LOAD_ASYNC_TO_LDS_B128 (ASYNCcnt) when available.
// B matrices kept [N][K] row-major so A and B fragments are contiguous-K loads.
// ---------------------------------------------------------------------------

#define LORA_SCALING 2.0f

typedef __attribute__((ext_vector_type(16))) __bf16 v16bf;
typedef __attribute__((ext_vector_type(8)))  float  v8f;
typedef __attribute__((ext_vector_type(4)))  float  f4;
typedef __attribute__((ext_vector_type(4)))  int    v4i;

union Frag {
    v16bf v;
    f4    f[2];
};

typedef unsigned short bf16_t;   // raw bf16 storage

// CDNA5 async global->LDS path (guarded: host pass / older toolchains fall back)
#if defined(__AMDGCN__) && __has_builtin(__builtin_amdgcn_global_load_async_to_lds_b128) && \
    __has_builtin(__builtin_amdgcn_s_wait_asynccnt)
#define USE_ASYNC_LDS 1
typedef __attribute__((address_space(1))) void as1_void;
typedef __attribute__((address_space(3))) void as3_void;
typedef __attribute__((address_space(1))) v4i  v4i_as1;
typedef __attribute__((address_space(3))) v4i  v4i_as3;
#else
#define USE_ASYNC_LDS 0
#endif

__device__ __forceinline__ bf16_t f32_to_bf16(float f) {
    union { float f; unsigned int u; } v; v.f = f;
    unsigned int u = v.u;
    u += 0x7FFFu + ((u >> 16) & 1u);     // round-to-nearest-even
    return (bf16_t)(u >> 16);
}

// ---------------------------------------------------------------------------
// Prep kernels
// ---------------------------------------------------------------------------

// h = x + (prev_token - x) * x_k, stored bf16 [N][C]
__global__ void k_token_shift(const float* __restrict__ x,
                              const float* __restrict__ x_prev,
                              const float* __restrict__ x_k,
                              bf16_t* __restrict__ h,
                              int B, int T, int C) {
    size_t i = (size_t)blockIdx.x * blockDim.x + threadIdx.x;
    size_t total = (size_t)B * T * C;
    if (i >= total) return;
    int c = (int)(i % C);
    size_t bt = i / C;
    int t = (int)(bt % T);
    float xv = x[i];
    float pv = (t == 0) ? x_prev[(bt / (size_t)T) * C + c] : x[i - C];
    h[i] = f32_to_bf16(xv + (pv - xv) * x_k[c]);
}

__global__ void k_convert(const float* __restrict__ src, bf16_t* __restrict__ dst, size_t n) {
    size_t i = (size_t)blockIdx.x * blockDim.x + threadIdx.x;
    if (i < n) dst[i] = f32_to_bf16(src[i]);
}

// src [rows][cols] fp32  ->  dst [cols][rows] bf16
__global__ void k_transpose_convert(const float* __restrict__ src, bf16_t* __restrict__ dst,
                                    int rows, int cols) {
    size_t i = (size_t)blockIdx.x * blockDim.x + threadIdx.x;
    size_t total = (size_t)rows * cols;
    if (i >= total) return;
    size_t r = i / cols;
    int c = (int)(i % cols);
    dst[(size_t)c * rows + r] = f32_to_bf16(src[i]);
}

__global__ void k_zero_f32(float* __restrict__ p, size_t n) {
    size_t i = (size_t)blockIdx.x * blockDim.x + threadIdx.x;
    if (i < n) p[i] = 0.0f;
}

// second output: x[:, T-1, :]  ->  dst [B][C]
__global__ void k_copy_last(const float* __restrict__ x, float* __restrict__ dst,
                            int B, int T, int C) {
    size_t i = (size_t)blockIdx.x * blockDim.x + threadIdx.x;
    if (i >= (size_t)B * C) return;
    size_t b = i / C;
    int c = (int)(i % C);
    dst[i] = x[(b * T + (T - 1)) * (size_t)C + c];
}

// ---------------------------------------------------------------------------
// Router + top-2 softmax gating. One wave32 per token (8 tokens / 256-thr block).
// ---------------------------------------------------------------------------
__global__ __launch_bounds__(256)
void k_router_gate(const float* __restrict__ x, const float* __restrict__ x_prev,
                   const float* __restrict__ x_k, const float* __restrict__ Router,
                   float* __restrict__ gating, int B, int T, int C) {
    int wave = threadIdx.x >> 5;
    int lane = threadIdx.x & 31;
    int n = blockIdx.x * 8 + wave;
    if (n >= B * T) return;
    int b = n / T, t = n % T;
    const float* xr = x + (size_t)n * C;
    const float* pr = (t == 0) ? (x_prev + (size_t)b * C) : (xr - C);
    float s1 = 0.f, s2 = 0.f, s3 = 0.f;
    for (int c = lane; c < C; c += 32) {
        float xv = xr[c];
        float hv = xv + (pr[c] - xv) * x_k[c];
        s1 += hv * Router[c];
        s2 += hv * Router[C + c];
        s3 += hv * Router[2 * C + c];
    }
    #pragma unroll
    for (int off = 16; off; off >>= 1) {
        s1 += __shfl_xor(s1, off, 32);
        s2 += __shfl_xor(s2, off, 32);
        s3 += __shfl_xor(s3, off, 32);
    }
    if (lane == 0) {
        float s[4] = {0.f, s1, s2, s3};
        int i0 = 0;
        for (int e = 1; e < 4; ++e) if (s[e] > s[i0]) i0 = e;       // first max wins ties
        int i1 = -1;
        for (int e = 0; e < 4; ++e) {
            if (e == i0) continue;
            if (i1 < 0 || s[e] > s[i1]) i1 = e;
        }
        float m  = s[i0];
        float e0 = __expf(s[i0] - m);
        float e1 = __expf(s[i1] - m);
        float inv = 1.f / (e0 + e1);
        float g[4] = {0.f, 0.f, 0.f, 0.f};
        g[i0] = e0 * inv;
        g[i1] = e1 * inv;
        float* gp = gating + (size_t)n * 4;
        gp[0] = g[0]; gp[1] = g[1]; gp[2] = g[2]; gp[3] = g[3];
    }
}

// ---------------------------------------------------------------------------
// WMMA GEMM:  acc += A[M][K] * B[N][K]^T
// Double-buffered LDS, BK = 64 (two WMMA K-chunks per staged tile).
//
// Fragment layouts per CDNA5 ISA (wave32, 16-bit operands):
//   A 16x32 : lanes 0-15 row M=lane   halves -> K 0..7,16..23
//             lanes16-31 row M=lane-16 halves -> K 8..15,24..31
//   B 32x16 : lanes 0-15 col N=lane   halves -> K 0..7,16..23
//             lanes16-31 col N=lane-16 halves -> K 8..15,24..31
// ---------------------------------------------------------------------------
template<int BM, int BN, int WAVE_M, int WAVE_N, int TM, int TN>
__device__ __forceinline__ void gemm_kphase(
    const bf16_t* __restrict__ A, size_t lda,
    const bf16_t* __restrict__ B, size_t ldb,
    int K,                     // multiple of 64
    size_t rowBlk, size_t colBlk,
    bf16_t* sA, bf16_t* sB,    // double buffers: 2*BM*64 and 2*BN*64 halves
    v8f (&acc)[TM][TN])
{
    const int tid    = threadIdx.x;
    const int waveId = tid >> 5;
    const int lane   = tid & 31;
    const int wm     = waveId / WAVE_N;
    const int wn     = waveId % WAVE_N;
    const int khalfA = (lane & 16) ? 8  : 0;   // A: K split 0/8
    const int kbaseB = (lane & 16) ? 16 : 0;   // unused: B also splits K 0/8 via khalfA
    const int lrow   = lane & 15;
    const int nk     = K / 64;

    // Stage one BK=64 tile pair into LDS buffer `buf`.
    auto stage = [&](int k0, int buf) {
        bf16_t* dA = sA + buf * (BM * 64);
        bf16_t* dB = sB + buf * (BN * 64);
#if USE_ASYNC_LDS
        #pragma unroll
        for (int idx = tid; idx < BM * 8; idx += 256) {     // 16B per lane per issue
            int r = idx >> 3, c8 = (idx & 7) * 8;
            __builtin_amdgcn_global_load_async_to_lds_b128(
                (v4i_as1*)(as1_void*)(const void*)&A[(rowBlk + r) * lda + k0 + c8],
                (v4i_as3*)(as3_void*)(void*)&dA[r * 64 + c8], 0, 0);
        }
        #pragma unroll
        for (int idx = tid; idx < BN * 8; idx += 256) {
            int r = idx >> 3, c8 = (idx & 7) * 8;
            __builtin_amdgcn_global_load_async_to_lds_b128(
                (v4i_as1*)(as1_void*)(const void*)&B[(colBlk + r) * ldb + k0 + c8],
                (v4i_as3*)(as3_void*)(void*)&dB[r * 64 + c8], 0, 0);
        }
#else
        #pragma unroll
        for (int idx = tid; idx < BM * 8; idx += 256) {
            int r = idx >> 3, c8 = (idx & 7) * 8;
            *(f4*)&dA[r * 64 + c8] = *(const f4*)&A[(rowBlk + r) * lda + k0 + c8];
        }
        #pragma unroll
        for (int idx = tid; idx < BN * 8; idx += 256) {
            int r = idx >> 3, c8 = (idx & 7) * 8;
            *(f4*)&dB[r * 64 + c8] = *(const f4*)&B[(colBlk + r) * ldb + k0 + c8];
        }
#endif
    };

    stage(0, 0);
    int cur = 0;
    for (int kt = 0; kt < nk; ++kt) {
#if USE_ASYNC_LDS
        __builtin_amdgcn_s_wait_asynccnt(0);   // my async writes to LDS landed
#endif
        __syncthreads();                       // everyone's writes visible; prior reads done
        if (kt + 1 < nk) stage((kt + 1) * 64, cur ^ 1);   // overlap with compute below

        const bf16_t* cA = sA + cur * (BM * 64);
        const bf16_t* cB = sB + cur * (BN * 64);
        #pragma unroll
        for (int kc = 0; kc < 64; kc += 32) {
            Frag a[TM], b[TN];
            #pragma unroll
            for (int i = 0; i < TM; ++i) {
                int r = wm * (BM / WAVE_M) + i * 16 + lrow;
                a[i].f[0] = *(const f4*)&cA[r * 64 + kc + khalfA];
                a[i].f[1] = *(const f4*)&cA[r * 64 + kc + 16 + khalfA];
            }
            #pragma unroll
            for (int j = 0; j < TN; ++j) {
                int r = wn * (BN / WAVE_N) + j * 16 + lrow;
                b[j].f[0] = *(const f4*)&cB[r * 64 + kc + khalfA];
                b[j].f[1] = *(const f4*)&cB[r * 64 + kc + 16 + khalfA];
            }
            #pragma unroll
            for (int i = 0; i < TM; ++i)
                #pragma unroll
                for (int j = 0; j < TN; ++j)
                    acc[i][j] = __builtin_amdgcn_wmma_f32_16x16x32_bf16(
                        false, a[i].v, false, b[j].v, (short)0, acc[i][j], false, false);
        }
        cur ^= 1;
    }
    __syncthreads();   // LDS free for the next phase / reuse
}

enum { EPI_RELU2 = 0, EPI_SCALE2 = 1, EPI_GATEACC = 2 };

// D = A1*B1^T (+ A2*B2^T)  with fused epilogue.
//   EPI_RELU2  : out_bf16 = bf16(relu(acc)^2)                  (k_e producer)
//   EPI_SCALE2 : out_bf16 = bf16(2 * acc)                      (LoRA rank-64 proj)
//   EPI_GATEACC: out_f32 += gating[row][expert] * acc          (final combine)
template<int BM, int BN, int WAVE_M, int WAVE_N, int EPI>
__global__ __launch_bounds__(256)
void k_gemm(const bf16_t* __restrict__ A1, size_t lda1,
            const bf16_t* __restrict__ B1, size_t ldb1, int K1,
            const bf16_t* __restrict__ A2, size_t lda2,
            const bf16_t* __restrict__ B2, size_t ldb2, int K2,
            bf16_t* __restrict__ outB, float* __restrict__ outF, size_t ldo,
            const float* __restrict__ gating, int expert)
{
    constexpr int TM = BM / WAVE_M / 16;
    constexpr int TN = BN / WAVE_N / 16;
    __shared__ __align__(16) bf16_t sA[2 * BM * 64];
    __shared__ __align__(16) bf16_t sB[2 * BN * 64];

    const size_t rowBlk = (size_t)blockIdx.y * BM;
    const size_t colBlk = (size_t)blockIdx.x * BN;

    v8f acc[TM][TN] = {};
    gemm_kphase<BM, BN, WAVE_M, WAVE_N>(A1, lda1, B1, ldb1, K1, rowBlk, colBlk, sA, sB, acc);
    if (K2 > 0)
        gemm_kphase<BM, BN, WAVE_M, WAVE_N>(A2, lda2, B2, ldb2, K2, rowBlk, colBlk, sA, sB, acc);

    // Epilogue. C/D layout: VGPR i -> (M = i [+8 for lanes>=16], N = lane&15).
    const int waveId = threadIdx.x >> 5;
    const int lane   = threadIdx.x & 31;
    const int wm     = waveId / WAVE_N;
    const int wn     = waveId % WAVE_N;
    const int rOff   = (lane & 16) ? 8 : 0;
    const int cOff   = lane & 15;

    #pragma unroll
    for (int i = 0; i < TM; ++i) {
        #pragma unroll
        for (int j = 0; j < TN; ++j) {
            size_t r0 = rowBlk + wm * (BM / WAVE_M) + i * 16 + rOff;
            size_t c0 = colBlk + wn * (BN / WAVE_N) + j * 16 + cOff;
            #pragma unroll
            for (int k = 0; k < 8; ++k) {
                size_t r = r0 + k;
                float v = acc[i][j][k];
                if constexpr (EPI == EPI_RELU2) {
                    float rv = v > 0.f ? v : 0.f;
                    outB[r * ldo + c0] = f32_to_bf16(rv * rv);
                } else if constexpr (EPI == EPI_SCALE2) {
                    outB[r * ldo + c0] = f32_to_bf16(LORA_SCALING * v);
                } else {
                    outF[r * ldo + c0] += gating[r * 4 + expert] * v;
                }
            }
        }
    }
}

// ---------------------------------------------------------------------------
// Host orchestration
// ---------------------------------------------------------------------------
extern "C" void kernel_launch(void* const* d_in, const int* in_sizes, int n_in,
                              void* d_out, int out_size, void* d_ws, size_t ws_size,
                              hipStream_t stream) {
    (void)in_sizes; (void)n_in; (void)out_size; (void)ws_size;

    const float* x       = (const float*)d_in[0];   // [B,T,C]
    const float* x_prev  = (const float*)d_in[1];   // [B,C]
    const float* x_k     = (const float*)d_in[2];   // [1,1,C]
    const float* Router  = (const float*)d_in[3];   // [E-1,C]
    const float* K_ref   = (const float*)d_in[4];   // [C,H]
    const float* V_ref   = (const float*)d_in[5];   // [H,C]
    const float* Ka      = (const float*)d_in[6];   // [E,R,C]
    const float* Kb      = (const float*)d_in[7];   // [E,H,R]
    const float* Va      = (const float*)d_in[8];   // [E,R,H]
    const float* Vb      = (const float*)d_in[9];   // [E,C,R]
    // d_in[10] = MoETopk (2), d_in[11] = MoECount (4): fixed for this problem.

    const int B = 4, T = 2048, C = 2048, H = 8192, R = 64, E = 4;
    const int N = B * T;

    // workspace carve-up (256B-aligned)
    char* ws = (char*)d_ws;
    auto alloc = [&](size_t bytes) -> char* {
        char* p = ws;
        ws += (bytes + 255) & ~(size_t)255;
        return p;
    };
    bf16_t* h_bf   = (bf16_t*)alloc((size_t)N * C * 2);     // h, bf16 [N][C]
    bf16_t* KrefT  = (bf16_t*)alloc((size_t)H * C * 2);     // K_ref^T  [H][C]
    bf16_t* VrefT  = (bf16_t*)alloc((size_t)C * H * 2);     // V_ref^T  [C][H]
    bf16_t* Ka_bf  = (bf16_t*)alloc((size_t)E * R * C * 2); // [E][R][C]
    bf16_t* Kb_bf  = (bf16_t*)alloc((size_t)E * H * R * 2); // [E][H][R]
    bf16_t* Va_bf  = (bf16_t*)alloc((size_t)E * R * H * 2); // [E][R][H]
    bf16_t* Vb_bf  = (bf16_t*)alloc((size_t)E * C * R * 2); // [E][C][R]
    float*  gating = (float*)alloc((size_t)N * E * 4);      // [N][E]
    bf16_t* a_bf   = (bf16_t*)alloc((size_t)N * R * 2);     // 2*(h @ Ka^T)
    bf16_t* b_bf   = (bf16_t*)alloc((size_t)N * R * 2);     // 2*(k @ Va^T)
    bf16_t* ke_bf  = (bf16_t*)alloc((size_t)N * H * 2);     // relu^2 k_e [N][H]

    float* outF  = (float*)d_out;            // [N][C]
    float* outXL = outF + (size_t)N * C;     // [B][C]

    const int THR = 256;
    auto blocks = [&](size_t n) { return (unsigned)((n + THR - 1) / THR); };

    // ---- prep ----
    k_token_shift<<<blocks((size_t)N * C), THR, 0, stream>>>(x, x_prev, x_k, h_bf, B, T, C);
    k_transpose_convert<<<blocks((size_t)C * H), THR, 0, stream>>>(K_ref, KrefT, C, H);
    k_transpose_convert<<<blocks((size_t)H * C), THR, 0, stream>>>(V_ref, VrefT, H, C);
    k_convert<<<blocks((size_t)E * R * C), THR, 0, stream>>>(Ka, Ka_bf, (size_t)E * R * C);
    k_convert<<<blocks((size_t)E * H * R), THR, 0, stream>>>(Kb, Kb_bf, (size_t)E * H * R);
    k_convert<<<blocks((size_t)E * R * H), THR, 0, stream>>>(Va, Va_bf, (size_t)E * R * H);
    k_convert<<<blocks((size_t)E * C * R), THR, 0, stream>>>(Vb, Vb_bf, (size_t)E * C * R);
    k_router_gate<<<N / 8, THR, 0, stream>>>(x, x_prev, x_k, Router, gating, B, T, C);
    k_zero_f32<<<blocks((size_t)N * C), THR, 0, stream>>>(outF, (size_t)N * C);
    k_copy_last<<<blocks((size_t)B * C), THR, 0, stream>>>(x, outXL, B, T, C);

    // ---- experts ----
    dim3 gLora(R / 64, N / 128);   // (1, 64)
    dim3 gK(H / 128, N / 128);     // (64, 64)
    dim3 gV(C / 128, N / 128);     // (16, 64)

    for (int e = 0; e < E; ++e) {
        // a = 2 * (h @ Ka[e]^T)                         [N][R]
        k_gemm<128, 64, 8, 1, EPI_SCALE2><<<gLora, THR, 0, stream>>>(
            h_bf, (size_t)C, Ka_bf + (size_t)e * R * C, (size_t)C, C,
            nullptr, 0, nullptr, 0, 0,
            a_bf, nullptr, (size_t)R, nullptr, 0);

        // k_e = relu(h @ K_ref + a @ Kb[e]^T)^2          [N][H]
        k_gemm<128, 128, 4, 2, EPI_RELU2><<<gK, THR, 0, stream>>>(
            h_bf, (size_t)C, KrefT, (size_t)C, C,
            a_bf, (size_t)R, Kb_bf + (size_t)e * H * R, (size_t)R, R,
            ke_bf, nullptr, (size_t)H, nullptr, 0);

        // b = 2 * (k_e @ Va[e]^T)                        [N][R]
        k_gemm<128, 64, 8, 1, EPI_SCALE2><<<gLora, THR, 0, stream>>>(
            ke_bf, (size_t)H, Va_bf + (size_t)e * R * H, (size_t)H, H,
            nullptr, 0, nullptr, 0, 0,
            b_bf, nullptr, (size_t)R, nullptr, 0);

        // out += gate[:,e] * (k_e @ V_ref + b @ Vb[e]^T) [N][C]
        k_gemm<128, 128, 4, 2, EPI_GATEACC><<<gV, THR, 0, stream>>>(
            ke_bf, (size_t)H, VrefT, (size_t)H, H,
            b_bf, (size_t)R, Vb_bf + (size_t)e * C * R, (size_t)R, R,
            nullptr, outF, (size_t)C, gating, e);
    }
}
